// ReferenceLocalAttention2d_46110768890406
// MI455X (gfx1250) — matrix operand order repaired and checked by reading, not compiled, not run on hardware
//
#include <hip/hip_runtime.h>

// ---- CDNA5 WMMA vector types -------------------------------------------------
typedef __attribute__((ext_vector_type(16))) __bf16 v16bf;
typedef __attribute__((ext_vector_type(8)))  __bf16 v8bf;
typedef __attribute__((ext_vector_type(8)))  float  v8f;

// ---- problem constants -------------------------------------------------------
#define HEADS 8
#define HD    32          // head dim == WMMA K for bf16 16x16x32
#define IMG   64
#define WIN   7
#define PAD   3

// ---- tiling ------------------------------------------------------------------
#define BY 8              // pixel rows per workgroup
#define BX 8              // pixel cols per workgroup
#define KY (BY + 6)       // 14 halo rows
#define KX 16             // 8 + 6 halo cols, padded to 16 -> wave key window is
                          // exactly 8x16 = 128 keys, contiguous per row in LDS
#define NKEY (KY * KX)    // 224
#define NWAVE 4           // 2x8 query tile per wave
#define NTHREADS (NWAVE * 32)

// LDS strides (bf16 elements unless noted); chosen 16B-aligned, bank-friendly
#define SKSTR 40          // sK  row stride  (80B = odd multiple of 16B)
#define SQSTR 40          // sQ  row stride
#define SVSTR 232         // sVt row stride  (464B)
#define SASTR 136         // sAttn row stride (272B)
#define SOSTR 33          // sOut row stride (floats)

static __device__ __forceinline__ v16bf cat16(v8bf a, v8bf b) {
  return __builtin_shufflevector(a, b, 0,1,2,3,4,5,6,7,8,9,10,11,12,13,14,15);
}

__global__ __launch_bounds__(NTHREADS)
void natten2d_wmma_kernel(const float* __restrict__ q,
                          const float* __restrict__ k,
                          const float* __restrict__ v,
                          float* __restrict__ out) {
  __shared__ __align__(16) __bf16 sQ[BY * BX * SQSTR];        // 5120 B
  __shared__ __align__(16) __bf16 sK[NKEY * SKSTR];           // 17920 B
  __shared__ __align__(16) __bf16 sVt[HD * SVSTR];            // 14848 B  [dim][key]
  __shared__ __align__(16) __bf16 sAttn[NWAVE * 16 * SASTR];  // 17408 B
  __shared__ __align__(16) float  sOut[BY * BX * SOSTR];      //  8448 B
  // total ~62 KB (< 64 KB static cap; WGP has 320 KB)

  const int tid  = threadIdx.x;
  const int lane = tid & 31;
  const int wave = tid >> 5;

  const int tile = blockIdx.x;           // 0..63 : 8x8 tiles over 64x64 image
  const int bh   = blockIdx.y;           // 0..31 : b*HEADS + h
  const int by0  = (tile >> 3) * BY;
  const int bx0  = (tile & 7) * BX;
  const size_t planeBase = (size_t)bh * HD * IMG * IMG;  // channel bh*32 + d

  // ---- stage Q: [64 px][32 d] bf16, coalesced along x ----------------------
  for (int i = tid; i < BY * BX * HD; i += NTHREADS) {
    const int x = i & 7, y = (i >> 3) & 7, d = i >> 6;
    const float f = q[planeBase + (size_t)d * (IMG * IMG) + (by0 + y) * IMG + (bx0 + x)];
    sQ[(y * BX + x) * SQSTR + d] = (__bf16)f;
  }
  // ---- stage K and V^T over the 14x16 halo region (zero-padded OOB) --------
  for (int i = tid; i < NKEY * HD; i += NTHREADS) {
    const int rx = i & 15;
    const int j  = i >> 4;
    const int ry = j % KY;
    const int d  = j / KY;
    const int gy = by0 - PAD + ry;
    const int gx = bx0 - PAD + rx;
    const bool ok = (gy >= 0) & (gy < IMG) & (gx >= 0) & (gx < IMG);
    const size_t off = planeBase + (size_t)d * (IMG * IMG) + gy * IMG + gx;
    const float kf = ok ? k[off] : 0.0f;
    const float vf = ok ? v[off] : 0.0f;
    const int ki = ry * KX + rx;
    sK[ki * SKSTR + d]  = (__bf16)kf;
    sVt[d * SVSTR + ki] = (__bf16)vf;
  }
  __syncthreads();

  // ---- per-wave query tile: 2 rows x 8 cols --------------------------------
  const int qy0  = wave * 2;             // block-relative query row origin
  const int col  = lane & 15;            // N index / A row index
  const int hs   = lane >> 4;            // half-wave select
  const int dbas = hs ? 16 : 0;          // A-fragment K base (contiguous k = 16h + s)

  // A fragment (Q tile, 16x32 bf16): lane holds row=col, dims {dbas..dbas+15}
  v16bf aQ;
  {
    const int qi = col;
    const int py = qy0 + (qi >> 3), px = qi & 7;
    const __bf16* qrow = &sQ[(py * BX + px) * SQSTR];
    v8bf lo = *(const v8bf*)(qrow + dbas);
    v8bf hi = *(const v8bf*)(qrow + dbas + 8);
    aQ = cat16(lo, hi);
  }

  // ---- GEMM1: scores[16q x 128keys] = Q(16x32) * K^T, 8 WMMAs --------------
  v8f c[8];
#pragma unroll
  for (int t = 0; t < 8; ++t) {
    const int ki = (qy0 + t) * KX + col;              // B: N=col, K-rows of key ki
    const __bf16* krow = &sK[ki * SKSTR + hs * 16];   // dims {16*hs .. 16*hs+15}
    v8bf lo = *(const v8bf*)(krow);
    v8bf hi = *(const v8bf*)(krow + 8);
    const v16bf bK = cat16(lo, hi);
    v8f z = {};
    c[t] = __builtin_amdgcn_wmma_f32_16x16x32_bf16(false, aQ, false, bK,
                                                   (short)0, z, false, false);
  }

  // ---- masked softmax over 128 local keys (49 valid window positions) ------
  const float scale = 0.17677669529663687f;   // 1/sqrt(32)
  const int gkx = bx0 + col - PAD;            // global key x for this lane
  const bool okx = (gkx >= 0) & (gkx < IMG);
#pragma unroll
  for (int r = 0; r < 8; ++r) {
    const int qi = r + 8 * hs;                // C layout: M = r + 8*(lane>>4)
    const int ty = qi >> 3, tx = qi & 7;
    float sv[8];
    float m = -1e30f;
#pragma unroll
    for (int t = 0; t < 8; ++t) {
      const int dy = t - ty;                  // window row offset 0..6 valid
      const int dx = col - tx;                // window col offset 0..6 valid
      const int gky = by0 + qy0 + t - PAD;
      const bool valid = (dy >= 0) & (dy < WIN) & (dx >= 0) & (dx < WIN) &
                         (gky >= 0) & (gky < IMG) & okx;
      const float s = valid ? c[t][r] * scale : -1e30f;
      sv[t] = s;
      m = fmaxf(m, s);
    }
#pragma unroll
    for (int off = 1; off < 16; off <<= 1) m = fmaxf(m, __shfl_xor(m, off, 32));
    float sum = 0.0f;
#pragma unroll
    for (int t = 0; t < 8; ++t) {
      const float p = (sv[t] > -5e29f) ? __expf(sv[t] - m) : 0.0f;
      sv[t] = p;
      sum += p;
    }
#pragma unroll
    for (int off = 1; off < 16; off <<= 1) sum += __shfl_xor(sum, off, 32);
    const float inv = 1.0f / sum;             // center key always valid -> sum > 0
#pragma unroll
    for (int t = 0; t < 8; ++t)
      sAttn[(wave * 16 + qi) * SASTR + 16 * t + col] = (__bf16)(sv[t] * inv);
  }

  // ---- GEMM2: out[16q x 32d] = attn(16x128) * V(128x32), 8 WMMAs -----------
  v8f o0 = {}, o1 = {};
#pragma unroll
  for (int cc = 0; cc < 4; ++cc) {
    // A fragment from sAttn: row=col, keys {32cc+dbas..32cc+dbas+15}
    const __bf16* arow = &sAttn[(wave * 16 + col) * SASTR + 32 * cc + dbas];
    v8bf alo = *(const v8bf*)(arow);
    v8bf ahi = *(const v8bf*)(arow + 8);
    const v16bf aA = cat16(alo, ahi);
    // B fragments from sVt: N=dim, K = 16 consecutive local keys (one halo row)
    const int ki0 = (qy0 + 2 * cc + hs) * KX;          // 16B-aligned (KX%8==0)
    {
      const __bf16* vrow = &sVt[col * SVSTR + ki0];     // dims 0..15
      v8bf blo = *(const v8bf*)(vrow);
      v8bf bhi = *(const v8bf*)(vrow + 8);
      o0 = __builtin_amdgcn_wmma_f32_16x16x32_bf16(false, aA, false, cat16(blo, bhi),
                                                   (short)0, o0, false, false);
    }
    {
      const __bf16* vrow = &sVt[(16 + col) * SVSTR + ki0]; // dims 16..31
      v8bf blo = *(const v8bf*)(vrow);
      v8bf bhi = *(const v8bf*)(vrow + 8);
      o1 = __builtin_amdgcn_wmma_f32_16x16x32_bf16(false, aA, false, cat16(blo, bhi),
                                                   (short)0, o1, false, false);
    }
  }

  // ---- transpose through LDS so global stores are contiguous ---------------
#pragma unroll
  for (int r = 0; r < 8; ++r) {
    const int qi = r + 8 * hs;
    const int py = qy0 + (qi >> 3), px = qi & 7;
    sOut[(py * BX + px) * SOSTR + col]      = o0[r];
    sOut[(py * BX + px) * SOSTR + 16 + col] = o1[r];
  }
  __syncthreads();

  // each thread stores two (y,d) rows of 8 contiguous floats (32B runs)
  {
    const int d  = tid & 31;
    const int y0 = tid >> 5;                 // 0..3 -> handles y0 and y0+4
#pragma unroll
    for (int yy = 0; yy < 2; ++yy) {
      const int y = y0 + 4 * yy;
      float* dst = out + planeBase + (size_t)d * (IMG * IMG) + (by0 + y) * IMG + bx0;
#pragma unroll
      for (int x = 0; x < BX; ++x)
        dst[x] = sOut[(y * BX + x) * SOSTR + d];
    }
  }
}

extern "C" void kernel_launch(void* const* d_in, const int* in_sizes, int n_in,
                              void* d_out, int out_size, void* d_ws, size_t ws_size,
                              hipStream_t stream) {
  const float* q = (const float*)d_in[0];
  const float* k = (const float*)d_in[1];
  const float* v = (const float*)d_in[2];
  float* out = (float*)d_out;
  (void)in_sizes; (void)n_in; (void)out_size; (void)d_ws; (void)ws_size;

  dim3 grid(64, 32);   // 64 8x8 pixel tiles  x  (B*HEADS)=32
  natten2d_wmma_kernel<<<grid, NTHREADS, 0, stream>>>(q, k, v, out);
}
